// Net_67491116089770
// MI455X (gfx1250) — matrix operand order repaired and checked by reading, not compiled, not run on hardware
//
#include <hip/hip_runtime.h>
#include <hip/hip_bf16.h>

typedef _Float16 f16;
typedef __attribute__((ext_vector_type(16))) _Float16 v16h;
typedef __attribute__((ext_vector_type(8)))  _Float16 v8h;
typedef __attribute__((ext_vector_type(8)))  float    v8f;

#define HH 512
#define WW 512

static __device__ __forceinline__ int refl(int j, int L) {
  return j < 0 ? -j : (j >= L ? 2 * L - 2 - j : j);
}
static __device__ __forceinline__ int imin(int a, int b) { return a < b ? a : b; }
static __device__ __forceinline__ int imax(int a, int b) { return a > b ? a : b; }
static __device__ __forceinline__ v8h zero8h() {
  v8h z;
#pragma unroll
  for (int j = 0; j < 8; ++j) z[j] = (_Float16)0.0f;
  return z;
}

// -------- weight packer: f32 OIHW (O,I,3,3) -> f16 [Tpad][16][Cpad], zero padded --------
__global__ void k_pack_w(const float* __restrict__ w, f16* __restrict__ dst,
                         int O, int I, int Cpad, int Tpad) {
  int idx = blockIdx.x * blockDim.x + threadIdx.x;
  int total = Tpad * 16 * Cpad;
  if (idx >= total) return;
  int c = idx % Cpad;
  int n = (idx / Cpad) % 16;
  int t = idx / (Cpad * 16);
  float v = 0.0f;
  if (t < 9 && n < O && c < I) v = w[((n * I + c) * 3 + t / 3) * 3 + (t % 3)];
  dst[idx] = (f16)v;
}

// -------- layer 0: 1x1 conv on 2x2 upsample of 1x1 input + coordconv concat (66 ch) --------
__global__ void k_layer0(const float* __restrict__ x, const float* __restrict__ w0,
                         const float* __restrict__ b0, f16* __restrict__ out) {
  int idx = blockIdx.x * blockDim.x + threadIdx.x;
  if (idx >= 3 * 2 * 2 * 66) return;
  int c  = idx % 66;
  int px = (idx / 66) % 2;
  int py = (idx / (66 * 2)) % 2;
  int n  = idx / (66 * 4);
  float v;
  if (c < 64) {
    v = b0[c];
#pragma unroll
    for (int ci = 0; ci < 3; ++ci) v += w0[c * 3 + ci] * x[n * 3 + ci];
    v = v > 0.0f ? v : 0.01f * v;
  } else if (c == 64) {
    v = 2.0f * (float)px;   // coordconv ch0: [[0,2],[0,2]]
  } else {
    v = 2.0f * (float)py;   // coordconv ch1: [[0,0],[2,2]]
  }
  out[idx] = (f16)v;        // NHWC [(n*2+py)*2+px]*66 + c
}

// -------- generic VALU conv for small layers: upsample x2 + reflect pad + 3x3 + lrelu ----
template <int INC, int P, int OUTC>
__global__ void k_conv_valu(const f16* __restrict__ in, const float* __restrict__ w,
                            const float* __restrict__ b, f16* __restrict__ out) {
  constexpr int R = 2 * P;
  constexpr int TOTAL = 3 * R * R * OUTC;
  int idx = blockIdx.x * blockDim.x + threadIdx.x;
  if (idx >= TOTAL) return;
  int co = idx % OUTC;
  int x  = (idx / OUTC) % R;
  int y  = (idx / (OUTC * R)) % R;
  int n  = idx / (OUTC * R * R);
  float acc = b[co];
#pragma unroll
  for (int ky = 0; ky < 3; ++ky) {
    int py = refl(y + ky - 1, R) >> 1;
#pragma unroll
    for (int kx = 0; kx < 3; ++kx) {
      int px = refl(x + kx - 1, R) >> 1;
      const f16*   ip = in + ((n * P + py) * P + px) * INC;
      const float* wp = w + co * INC * 9 + ky * 3 + kx;
      for (int ci = 0; ci < INC; ++ci) acc += (float)ip[ci] * wp[ci * 9];
    }
  }
  acc = acc > 0.0f ? acc : 0.01f * acc;
  out[idx] = (f16)acc;
}

// -------- WMMA conv, C_in = 16. FINAL: no upsample, tanh, store f32 flows (6 ch) --------
// M = 16 output pixels along x, N = 16 output channels, K = 2 taps x 16 ch per WMMA.
template <bool FINAL, int P>
__global__ void k_conv_wmma16(const f16* __restrict__ in,
                              const f16* __restrict__ wpk, const float* __restrict__ bias,
                              f16* __restrict__ out, float* __restrict__ flows) {
  constexpr int R      = FINAL ? P : (P << 1);
  constexpr int INRES  = FINAL ? R : P;
  constexpr int TILESX = R >> 4;
  constexpr int NTILES = 3 * R * TILESX;
  int wave = (int)((blockIdx.x * blockDim.x + threadIdx.x) >> 5);
  int lane = (int)(threadIdx.x & 31);
  if (wave >= NTILES) return;                // uniform per wave: EXEC stays all-ones
  int tx = wave % TILESX;
  int y  = (wave / TILESX) % R;
  int b  = wave / (TILESX * R);
  int x0 = tx << 4;
  int m = lane & 15, hi = lane >> 4;
  int x = x0 + m;
  int n = lane & 15;

  auto loadA = [&](int t, int c0) -> v8h {   // 16B contiguous A-fragment half
    int ky = t / 3, kx = t % 3;
    int py = refl(y + ky - 1, R), px = refl(x + kx - 1, R);
    if (!FINAL) { py >>= 1; px >>= 1; }
    return *(const v8h*)(in + (((b * INRES + py) * INRES + px) << 4) + c0);
  };

  v8f acc = {};
  // k-steps 0..3 (taps 0..7): cap unroll to limit loads in flight
#pragma unroll 2
  for (int s = 0; s < 4; ++s) {
    v8h alo = loadA(2 * s + hi, 0);
    v8h ahi = loadA(2 * s + hi, 8);
    v16h A;
#pragma unroll
    for (int j = 0; j < 8; ++j) { A[j] = alo[j]; A[8 + j] = ahi[j]; }
    int t = 2 * s + hi;                      // wpk: [tap][n][c16], 10 taps (tap 9 zero)
    v16h B = *(const v16h*)(wpk + ((t << 4) + n) * 16);
    acc = __builtin_amdgcn_wmma_f32_16x16x32_f16(false, A, false, B, (short)0, acc,
                                                 false, false);
  }
  { // k-step 4: tap 8 + zero K-padding
    v8h alo = loadA(8, 0);
    v8h ahi = loadA(8, 8);
    v16h A;
#pragma unroll
    for (int j = 0; j < 8; ++j) { A[j] = alo[j]; A[8 + j] = ahi[j]; }
    int t = 8 + hi;                          // hi=1 -> tap 9 (zeroed)
    v16h B = *(const v16h*)(wpk + ((t << 4) + n) * 16);
    acc = __builtin_amdgcn_wmma_f32_16x16x32_f16(false, A, false, B, (short)0, acc,
                                                 false, false);
  }

  float bv = 0.0f;
  if (!FINAL || n < 6) bv = bias[n];
#pragma unroll
  for (int r = 0; r < 8; ++r) {
    int mm = hi * 8 + r;                     // D layout: VGPR r, hi selects M+8
    float v = acc[r] + bv;
    if constexpr (FINAL) {
      if (n < 6) flows[((b * 6 + n) * HH + y) * WW + (x0 + mm)] = tanhf(v);
    } else {
      v = v > 0.0f ? v : 0.01f * v;
      out[(((b * R + y) * R + (x0 + mm)) << 4) + n] = (f16)v;
    }
  }
}

// -------- WMMA conv, C_in = 32 (layer 6), one tap per K-step --------
template <int P>
__global__ void k_conv_wmma32(const f16* __restrict__ in,
                              const f16* __restrict__ wpk, const float* __restrict__ bias,
                              f16* __restrict__ out) {
  constexpr int R      = P << 1;
  constexpr int TILESX = R >> 4;
  constexpr int NTILES = 3 * R * TILESX;
  int wave = (int)((blockIdx.x * blockDim.x + threadIdx.x) >> 5);
  int lane = (int)(threadIdx.x & 31);
  if (wave >= NTILES) return;
  int tx = wave % TILESX;
  int y  = (wave / TILESX) % R;
  int b  = wave / (TILESX * R);
  int x0 = tx << 4;
  int m = lane & 15, hi = lane >> 4;
  int x = x0 + m;
  int n = lane & 15;

  v8f acc = {};
#pragma unroll 3
  for (int t = 0; t < 9; ++t) {
    int ky = t / 3, kx = t % 3;
    int py = refl(y + ky - 1, R) >> 1;
    int px = refl(x + kx - 1, R) >> 1;
    const f16* ip = in + (((b * P + py) * P + px) << 5);  // 32 ch/pixel
    v8h alo = *(const v8h*)(ip + hi * 16);                // K = c in [hi*16, hi*16+8)
    v8h ahi = *(const v8h*)(ip + hi * 16 + 8);            // K = c in [hi*16+8, hi*16+16)
    v16h A;
#pragma unroll
    for (int j = 0; j < 8; ++j) { A[j] = alo[j]; A[8 + j] = ahi[j]; }
    v16h B = *(const v16h*)(wpk + ((t << 4) + n) * 32 + hi * 16);
    acc = __builtin_amdgcn_wmma_f32_16x16x32_f16(false, A, false, B, (short)0, acc,
                                                 false, false);
  }
  float bv = bias[n];
#pragma unroll
  for (int r = 0; r < 8; ++r) {
    int mm = hi * 8 + r;
    float v = acc[r] + bv;
    v = v > 0.0f ? v : 0.01f * v;
    out[(((b * R + y) * R + (x0 + mm)) << 4) + n] = (f16)v;
  }
}

// -------- blending: per-pixel bilinear warps + weighting --------
// Neighbor loop kept serialized (unroll 1) to halve live ranges and avoid scratch spill.
__global__ void k_blend(const float* __restrict__ xin, const float* __restrict__ nb,
                        const int* __restrict__ aidx, const float* __restrict__ albedos,
                        const float* __restrict__ flows, float* __restrict__ out) {
  int p = blockIdx.x * blockDim.x + threadIdx.x;
  const int HWc = HH * WW;
  if (p >= HWc) return;
  int y = p / WW, x = p % WW;
  const float* alb = albedos + (size_t)aidx[0] * 3 * HWc;

  float f0[6];
#pragma unroll
  for (int c = 0; c < 6; ++c) f0[c] = flows[c * HWc + p];
  float bx = -1.0f + 2.0f * (float)x / (float)(WW - 1);
  float by = -1.0f + 2.0f * (float)y / (float)(HH - 1);
  float alh[3];
#pragma unroll
  for (int c = 0; c < 3; ++c) alh[c] = alb[c * HWc + p];

  float acc[3] = {0.0f, 0.0f, 0.0f};
  float wsum = 0.0f;
#pragma unroll 1
  for (int n = 0; n < 2; ++n) {
    float dl = xin[0] - xin[(1 + n) * 3 + 0];
    float dv = xin[1] - xin[(1 + n) * 3 + 1];
    float dt = xin[2] - xin[(1 + n) * 3 + 2];
    bool flag = fabsf(dl) > 0.0f;
    float offx = dl * f0[0] + dv * f0[2] + dt * f0[4];
    float offy = dl * f0[1] + dv * f0[3] + dt * f0[5];
    float gx = fminf(fmaxf((bx + offx + 1.0f) * (float)WW * 0.5f - 0.5f, 0.0f), (float)(WW - 1));
    float gy = fminf(fmaxf((by + offy + 1.0f) * (float)HH * 0.5f - 0.5f, 0.0f), (float)(HH - 1));
    float x0f = floorf(gx), y0f = floorf(gy);
    float wx = gx - x0f, wy = gy - y0f;
    int x0i = imin(imax((int)x0f, 0), WW - 1);
    int x1i = imin(x0i + 1, WW - 1);
    int y0i = imin(imax((int)y0f, 0), HH - 1);
    int y1i = imin(y0i + 1, HH - 1);
    int i00 = y0i * WW + x0i, i01 = y0i * WW + x1i;
    int i10 = y1i * WW + x0i, i11 = y1i * WW + x1i;
    float w00 = (1.0f - wx) * (1.0f - wy), w01 = wx * (1.0f - wy);
    float w10 = (1.0f - wx) * wy,          w11 = wx * wy;

    float ws3[3];
#pragma unroll
    for (int c = 0; c < 3; ++c) {
      const float* np_ = nb + (n * 3 + c) * HWc;
      const float* ap_ = alb + c * HWc;
      float s00 = flag ? np_[i00] / ap_[i00] : np_[i00];
      float s01 = flag ? np_[i01] / ap_[i01] : np_[i01];
      float s10 = flag ? np_[i10] / ap_[i10] : np_[i10];
      float s11 = flag ? np_[i11] / ap_[i11] : np_[i11];
      ws3[c] = w00 * s00 + w01 * s01 + w10 * s10 + w11 * s11;
    }
    float wlf[2], wvf[2], wtf[2];
#pragma unroll
    for (int c = 0; c < 2; ++c) {
      const float* lp = flows + ((1 + n) * 6 + 0 + c) * HWc;
      const float* vp = flows + ((1 + n) * 6 + 2 + c) * HWc;
      const float* tp = flows + ((1 + n) * 6 + 4 + c) * HWc;
      wlf[c] = w00 * lp[i00] + w01 * lp[i01] + w10 * lp[i10] + w11 * lp[i11];
      wvf[c] = w00 * vp[i00] + w01 * vp[i01] + w10 * vp[i10] + w11 * vp[i11];
      wtf[c] = w00 * tp[i00] + w01 * tp[i01] + w10 * tp[i10] + w11 * tp[i11];
    }
    float obx = dl * wlf[0] + dv * wvf[0] + dt * wtf[0];
    float oby = dl * wlf[1] + dv * wvf[1] + dt * wtf[1];
    float dist = fabsf(offx - obx) + fabsf(offy - oby);
    float wgt = expf(-0.1f * (float)WW * dist);
#pragma unroll
    for (int c = 0; c < 3; ++c) {
      float wi = flag ? ws3[c] * alh[c] : ws3[c];
      acc[c] += wi * wgt;
    }
    wsum += wgt;
  }
  float inv = 1.0f / (wsum + 1e-5f);
#pragma unroll
  for (int c = 0; c < 3; ++c) out[c * HWc + p] = acc[c] * inv;
}

static inline int cdiv(int a, int b) { return (a + b - 1) / b; }

extern "C" void kernel_launch(void* const* d_in, const int* in_sizes, int n_in,
                              void* d_out, int out_size, void* d_ws, size_t ws_size,
                              hipStream_t stream) {
  (void)in_sizes; (void)n_in; (void)out_size; (void)ws_size;
  const float* x         = (const float*)d_in[0];
  const float* neighbors = (const float*)d_in[1];
  const int*   aidx      = (const int*)d_in[2];
  const float* w[9]; const float* bb[9];
  for (int i = 0; i < 9; ++i) { w[i] = (const float*)d_in[3 + 2 * i]; bb[i] = (const float*)d_in[4 + 2 * i]; }
  const float* wf      = (const float*)d_in[21];
  const float* bf      = (const float*)d_in[22];
  const float* albedos = (const float*)d_in[23];
  float* out = (float*)d_out;

  char* ws = (char*)d_ws;
  size_t off = 0;
  auto alloc = [&](size_t bytes) -> void* {
    off = (off + 255) & ~(size_t)255;
    void* p = ws + off;
    off += bytes;
    return p;
  };
  f16* a0 = (f16*)alloc((size_t)3 * 2 * 2 * 66 * 2);
  f16* a1 = (f16*)alloc((size_t)3 * 4 * 4 * 64 * 2);
  f16* a2 = (f16*)alloc((size_t)3 * 8 * 8 * 64 * 2);
  f16* a3 = (f16*)alloc((size_t)3 * 16 * 16 * 32 * 2);
  f16* a4 = (f16*)alloc((size_t)3 * 32 * 32 * 32 * 2);
  f16* a5 = (f16*)alloc((size_t)3 * 64 * 64 * 32 * 2);
  f16* a6 = (f16*)alloc((size_t)3 * 128 * 128 * 16 * 2);
  f16* a7 = (f16*)alloc((size_t)3 * 256 * 256 * 16 * 2);
  f16* a8 = (f16*)alloc((size_t)3 * 512 * 512 * 16 * 2);
  float* flows = (float*)alloc((size_t)3 * 6 * 512 * 512 * 4);
  f16* wpk6 = (f16*)alloc((size_t)9 * 16 * 32 * 2);
  f16* wpk7 = (f16*)alloc((size_t)10 * 16 * 16 * 2);
  f16* wpk8 = (f16*)alloc((size_t)10 * 16 * 16 * 2);
  f16* wpkf = (f16*)alloc((size_t)10 * 16 * 16 * 2);

  const int B = 256;
  // weight packing for WMMA layers
  k_pack_w<<<cdiv(9 * 16 * 32, B), B, 0, stream>>>(w[6], wpk6, 16, 32, 32, 9);
  k_pack_w<<<cdiv(10 * 16 * 16, B), B, 0, stream>>>(w[7], wpk7, 16, 16, 16, 10);
  k_pack_w<<<cdiv(10 * 16 * 16, B), B, 0, stream>>>(w[8], wpk8, 16, 16, 16, 10);
  k_pack_w<<<cdiv(10 * 16 * 16, B), B, 0, stream>>>(wf,   wpkf,  6, 16, 16, 10);

  // small layers (VALU, compile-time dims)
  k_layer0<<<cdiv(3 * 2 * 2 * 66, B), B, 0, stream>>>(x, w[0], bb[0], a0);
  k_conv_valu<66, 2, 64><<<cdiv(3 * 4 * 4 * 64, B), B, 0, stream>>>(a0, w[1], bb[1], a1);
  k_conv_valu<64, 4, 64><<<cdiv(3 * 8 * 8 * 64, B), B, 0, stream>>>(a1, w[2], bb[2], a2);
  k_conv_valu<64, 8, 32><<<cdiv(3 * 16 * 16 * 32, B), B, 0, stream>>>(a2, w[3], bb[3], a3);
  k_conv_valu<32, 16, 32><<<cdiv(3 * 32 * 32 * 32, B), B, 0, stream>>>(a3, w[4], bb[4], a4);
  k_conv_valu<32, 32, 32><<<cdiv(3 * 64 * 64 * 32, B), B, 0, stream>>>(a4, w[5], bb[5], a5);

  // big layers (WMMA): waves exactly tile the work (blocks = tiles/8), EXEC all-ones
  k_conv_wmma32<64><<<3072 / 8, B, 0, stream>>>(a5, wpk6, bb[6], a6);                   // L6: 32->16 @128^2
  k_conv_wmma16<false, 128><<<12288 / 8, B, 0, stream>>>(a6, wpk7, bb[7], a7, nullptr); // L7: 16->16 @256^2
  k_conv_wmma16<false, 256><<<49152 / 8, B, 0, stream>>>(a7, wpk8, bb[8], a8, nullptr); // L8: 16->16 @512^2
  k_conv_wmma16<true, 512><<<49152 / 8, B, 0, stream>>>(a8, wpkf, bf, nullptr, flows);  // final: 16->6 + tanh

  // blending
  k_blend<<<cdiv(HH * WW, B), B, 0, stream>>>(x, neighbors, aidx, albedos, flows, out);
}
